// N_ARY_GEN_23691039605050
// MI455X (gfx1250) — matrix operand order repaired and checked by reading, not compiled, not run on hardware
//
#include <hip/hip_runtime.h>
#include <hip/hip_bf16.h>
#include <math.h>

// ---------------------------------------------------------------------------
// Problem constants (from reference)
// ---------------------------------------------------------------------------
#define D_DIM   768      // embedding dim
#define CD_DIM  256      // composed dim
#define H_DIM   1024     // MLP hidden (4*CD)
#define N_TOK   32768    // B*S
#define N_SPAN  4096     // spans per level
#define NC      4        // span width

// LDS row pads: row stride (in dwords) mod 64 banks == 4 -> conflict-free
#define LDA (D_DIM + 8)   // 776 halves = 1552B
#define LDM (CD_DIM + 8)  // 264 halves = 528B
#define LDH (H_DIM + 8)   // 1032 halves = 2064B

typedef __bf16 bf16_t;
typedef bf16_t bf16x8 __attribute__((ext_vector_type(8)));
typedef bf16_t v16bf  __attribute__((ext_vector_type(16)));
typedef float  v8f    __attribute__((ext_vector_type(8)));
typedef float  f32x4  __attribute__((ext_vector_type(4)));

// Build a 16-element bf16 fragment from two contiguous 8-half (16B) chunks.
__device__ __forceinline__ v16bf make_frag(const bf16_t* p0, const bf16_t* p1) {
    bf16x8 c0 = *(const bf16x8*)p0;
    bf16x8 c1 = *(const bf16x8*)p1;
    return __builtin_shufflevector(c0, c1, 0,1,2,3,4,5,6,7,8,9,10,11,12,13,14,15);
}

__device__ __forceinline__ v8f wmma_bf16(v16bf a, v16bf b, v8f c) {
    // v_wmma_f32_16x16x32_bf16  (8-arg form: neg_a, A, neg_b, B, c_mod, C, reuse_a, reuse_b)
    return __builtin_amdgcn_wmma_f32_16x16x32_bf16(false, a, false, b, (short)0, c,
                                                   false, false);
}

__device__ __forceinline__ float gelu_tanh(float x) {
    // jax.nn.gelu default (tanh approximation)
    const float k0 = 0.7978845608028654f, k1 = 0.044715f;
    return 0.5f * x * (1.0f + tanhf(k0 * (x + k1 * x * x * x)));
}

// ---------------------------------------------------------------------------
// Kernel W: convert + transpose learned weights into bf16, column-major
//   wdT [256][768], wc1T [1024][256], wc2T [256][1024]
// ---------------------------------------------------------------------------
#define WDT_ELEMS  (CD_DIM * D_DIM)   // 196608
#define WC1T_ELEMS (H_DIM * CD_DIM)   // 262144
#define WC2T_ELEMS (CD_DIM * H_DIM)   // 262144
#define PREP_TOTAL (WDT_ELEMS + WC1T_ELEMS + WC2T_ELEMS)  // 720896

__global__ __launch_bounds__(256) void prep_weights_kernel(
    const float* __restrict__ w_down, const float* __restrict__ wc1,
    const float* __restrict__ wc2,
    bf16_t* __restrict__ wdT, bf16_t* __restrict__ wc1T, bf16_t* __restrict__ wc2T)
{
    size_t i = (size_t)blockIdx.x * 256 + threadIdx.x;
    if (i < WDT_ELEMS) {
        int c = (int)(i / D_DIM), k = (int)(i % D_DIM);      // wdT[c][k] = w_down[k][c]
        wdT[i] = (bf16_t)w_down[(size_t)k * CD_DIM + c];
    } else if (i < (size_t)WDT_ELEMS + WC1T_ELEMS) {
        size_t j = i - WDT_ELEMS;
        int c = (int)(j / CD_DIM), k = (int)(j % CD_DIM);    // wc1T[c][k] = wc1[k][c]
        wc1T[j] = (bf16_t)wc1[(size_t)k * H_DIM + c];
    } else if (i < (size_t)PREP_TOTAL) {
        size_t j = i - WDT_ELEMS - WC1T_ELEMS;
        int c = (int)(j / H_DIM), k = (int)(j % H_DIM);      // wc2T[c][k] = wc2[k][c]
        wc2T[j] = (bf16_t)wc2[(size_t)k * CD_DIM + c];
    }
}

// ---------------------------------------------------------------------------
// Kernel A: fused embedding gather + down projection
//   out[n, c] = sum_d emb[ids[n], d] * w_down[d, c] + b_down[c]
// One block = 16 token rows x 256 cols, 8 waves, each wave -> 2 col-tiles.
// A (gathered emb rows, bf16) staged once in LDS; B from L2-resident wdT.
// ---------------------------------------------------------------------------
__global__ __launch_bounds__(256) void embed_down_kernel(
    const int* __restrict__ ids, const float* __restrict__ emb,
    const bf16_t* __restrict__ wdT, const float* __restrict__ b_down,
    float* __restrict__ out)
{
    __shared__ bf16_t ldsA[16 * LDA];
    const int row0 = blockIdx.x * 16;
    const int t = threadIdx.x;

    // Stage: 16 rows x 768 cols as float4 -> bf16 into LDS (12 iters/thread)
    for (int i = t; i < 16 * (D_DIM / 4); i += 256) {
        int r  = i / (D_DIM / 4);
        int c4 = i % (D_DIM / 4);
        int id = ids[row0 + r];
        if (id == -100) id = 0;
        f32x4 v = *(const f32x4*)(emb + (size_t)id * D_DIM + (size_t)c4 * 4);
        bf16_t* dst = &ldsA[r * LDA + c4 * 4];
        dst[0] = (bf16_t)v[0]; dst[1] = (bf16_t)v[1];
        dst[2] = (bf16_t)v[2]; dst[3] = (bf16_t)v[3];
    }
    __syncthreads();

    const int lane = t & 31;
    const int wave = t >> 5;
    const int m  = lane & 15;   // A row within tile
    const int h  = lane >> 4;   // A half (K-group select)
    const int n  = lane & 15;   // B/C column within tile
    const int kg = lane >> 4;   // B K-group select
    const int c0 = wave * 32;

    v8f acc0 = {}; v8f acc1 = {};
    for (int k0 = 0; k0 < D_DIM; k0 += 32) {
        const bf16_t* ab = &ldsA[m * LDA + k0];
        v16bf a = make_frag(ab + h * 16, ab + h * 16 + 8);
        const bf16_t* b0 = wdT + (size_t)(c0 + n)      * D_DIM + k0 + kg * 16;
        const bf16_t* b1 = wdT + (size_t)(c0 + 16 + n) * D_DIM + k0 + kg * 16;
        acc0 = wmma_bf16(a, make_frag(b0, b0 + 8), acc0);
        acc1 = wmma_bf16(a, make_frag(b1, b1 + 8), acc1);
    }

    const float bias0 = b_down[c0 + n];
    const float bias1 = b_down[c0 + 16 + n];
#pragma unroll
    for (int r = 0; r < 8; ++r) {
        int orow = row0 + r + h * 8;   // C layout: lanes 0-15 -> M=r, lanes 16-31 -> M=8+r
        out[(size_t)orow * CD_DIM + c0 + n]      = acc0[r] + bias0;
        out[(size_t)orow * CD_DIM + c0 + 16 + n] = acc1[r] + bias1;
    }
}

// ---------------------------------------------------------------------------
// Kernel B: one span level's compose: masked mean -> GELU MLP -> compbuf
// Block = 16 span rows. GEMM1: 16x256 @ 256x1024 (8 tiles/wave).
// GEMM2: 16x1024 @ 1024x256 (2 tiles/wave). h kept in LDS as bf16.
// ---------------------------------------------------------------------------
__global__ __launch_bounds__(256) void compose_kernel(
    const float* __restrict__ buf, const int* __restrict__ spans,
    const bf16_t* __restrict__ wc1T, const float* __restrict__ bc1,
    const bf16_t* __restrict__ wc2T, const float* __restrict__ bc2,
    float* __restrict__ compbuf)
{
    __shared__ bf16_t ldsM[16 * LDM];
    __shared__ bf16_t ldsH[16 * LDH];
    const int row0 = blockIdx.x * 16;
    const int t = threadIdx.x;

    // Masked-mean gather: 16 rows x 256 cols (16 iters/thread)
    for (int i = t; i < 16 * CD_DIM; i += 256) {
        int r = i >> 8;
        int c = i & 255;
        const int* sp = spans + (size_t)(row0 + r) * NC;
        float sum = 0.f, cnt = 0.f;
#pragma unroll
        for (int j = 0; j < NC; ++j) {
            int s = sp[j];
            if (s != -100) { sum += buf[(size_t)s * CD_DIM + c]; cnt += 1.f; }
        }
        ldsM[r * LDM + c] = (bf16_t)(sum / fmaxf(cnt, 1.0f));
    }
    __syncthreads();

    const int lane = t & 31;
    const int wave = t >> 5;
    const int m  = lane & 15;
    const int h  = lane >> 4;
    const int n  = lane & 15;
    const int kg = lane >> 4;

    // GEMM1 + GELU -> ldsH
    {
        v8f acc[8] = {};
        for (int k0 = 0; k0 < CD_DIM; k0 += 32) {
            const bf16_t* ab = &ldsM[m * LDM + k0];
            v16bf a = make_frag(ab + h * 16, ab + h * 16 + 8);
#pragma unroll
            for (int s = 0; s < 8; ++s) {
                int c0 = wave * 128 + s * 16;
                const bf16_t* bp = wc1T + (size_t)(c0 + n) * CD_DIM + k0 + kg * 16;
                acc[s] = wmma_bf16(a, make_frag(bp, bp + 8), acc[s]);
            }
        }
#pragma unroll
        for (int s = 0; s < 8; ++s) {
            int c0 = wave * 128 + s * 16;
            float bias = bc1[c0 + n];
#pragma unroll
            for (int r = 0; r < 8; ++r) {
                float x = acc[s][r] + bias;
                ldsH[(r + h * 8) * LDH + c0 + n] = (bf16_t)gelu_tanh(x);
            }
        }
    }
    __syncthreads();

    // GEMM2 -> compbuf
    {
        const int c0 = wave * 32;
        v8f acc0 = {}; v8f acc1 = {};
        for (int k0 = 0; k0 < H_DIM; k0 += 32) {
            const bf16_t* ab = &ldsH[m * LDH + k0];
            v16bf a = make_frag(ab + h * 16, ab + h * 16 + 8);
            const bf16_t* b0 = wc2T + (size_t)(c0 + n)      * H_DIM + k0 + kg * 16;
            const bf16_t* b1 = wc2T + (size_t)(c0 + 16 + n) * H_DIM + k0 + kg * 16;
            acc0 = wmma_bf16(a, make_frag(b0, b0 + 8), acc0);
            acc1 = wmma_bf16(a, make_frag(b1, b1 + 8), acc1);
        }
        const float bias0 = bc2[c0 + n];
        const float bias1 = bc2[c0 + 16 + n];
#pragma unroll
        for (int r = 0; r < 8; ++r) {
            int orow = row0 + r + h * 8;
            compbuf[(size_t)orow * CD_DIM + c0 + n]      = acc0[r] + bias0;
            compbuf[(size_t)orow * CD_DIM + c0 + 16 + n] = acc1[r] + bias1;
        }
    }
}

// ---------------------------------------------------------------------------
// Kernel C: scatter comp rows back into buf at max(spans)+1
// (separate launch so every compose gather sees the pre-scatter buffer)
// ---------------------------------------------------------------------------
__global__ __launch_bounds__(256) void scatter_kernel(
    const int* __restrict__ spans, const float* __restrict__ compbuf,
    float* __restrict__ buf)
{
    const int row = blockIdx.x;
    const int* sp = spans + (size_t)row * NC;
    int mx = sp[0];
    int s1 = sp[1]; mx = s1 > mx ? s1 : mx;
    int s2 = sp[2]; mx = s2 > mx ? s2 : mx;
    int s3 = sp[3]; mx = s3 > mx ? s3 : mx;
    const int tgt = mx + 1;
    buf[(size_t)tgt * CD_DIM + threadIdx.x] = compbuf[(size_t)row * CD_DIM + threadIdx.x];
}

// ---------------------------------------------------------------------------
// Launch
// ---------------------------------------------------------------------------
extern "C" void kernel_launch(void* const* d_in, const int* in_sizes, int n_in,
                              void* d_out, int out_size, void* d_ws, size_t ws_size,
                              hipStream_t stream) {
    (void)in_sizes; (void)n_in; (void)out_size; (void)ws_size;

    const int*   ids    = (const int*)d_in[0];
    const int*   spans0 = (const int*)d_in[1];
    const int*   spans1 = (const int*)d_in[2];
    const int*   spans2 = (const int*)d_in[3];
    // d_in[4] = opening_nt_start_idx (unused by reference body)
    const float* emb    = (const float*)d_in[5];
    const float* w_down = (const float*)d_in[6];
    const float* b_down = (const float*)d_in[7];
    const float* wc1    = (const float*)d_in[8];
    const float* bc1    = (const float*)d_in[9];
    const float* wc2    = (const float*)d_in[10];
    const float* bc2    = (const float*)d_in[11];
    float* out = (float*)d_out;   // out IS buf: [32768, 256] f32

    // Workspace layout (bytes): bf16 transposed weights + per-level comp rows
    char* ws = (char*)d_ws;
    bf16_t* wdT     = (bf16_t*)(ws + 0);                               // 393216 B
    bf16_t* wc1T    = (bf16_t*)(ws + 393216);                          // 524288 B
    bf16_t* wc2T    = (bf16_t*)(ws + 917504);                          // 524288 B
    float*  compbuf = (float*) (ws + 1441792);                         // 4 MiB

    // 1) weights -> bf16 column-major
    prep_weights_kernel<<<(PREP_TOTAL + 255) / 256, 256, 0, stream>>>(
        w_down, wc1, wc2, wdT, wc1T, wc2T);

    // 2) embedding gather + down projection -> buf
    embed_down_kernel<<<N_TOK / 16, 256, 0, stream>>>(ids, emb, wdT, b_down, out);

    // 3) three dependent span levels: compose then scatter
    const int* levels[3] = {spans0, spans1, spans2};
    for (int l = 0; l < 3; ++l) {
        compose_kernel<<<N_SPAN / 16, 256, 0, stream>>>(
            out, levels[l], wc1T, bc1, wc2T, bc2, compbuf);
        scatter_kernel<<<N_SPAN, 256, 0, stream>>>(levels[l], compbuf, out);
    }
}
